// CrystalGraphConv_77653008712169
// MI455X (gfx1250) — matrix operand order repaired and checked by reading, not compiled, not run on hardware
//
#include <hip/hip_runtime.h>
#include <stdint.h>
#include <stddef.h>

typedef __attribute__((ext_vector_type(16))) __bf16 v16bf;
typedef __attribute__((ext_vector_type(8)))  float  v8f;

#define BB   512
#define NN   64
#define MM   12
#define FF   64
#define NBRF 41
#define KDIM 169          // 2F + NF
#define KPAD 192          // 6 * 32 (K-steps of the bf16 WMMA)
#define KP   196          // padded LDS K-stride (bank-conflict avoidance)
#define COLS 128          // 2F output columns
#define RPB  128          // rows per block (8 waves x 16 rows)
#define RTOT (BB*NN*MM)   // 393216 rows, divisible by RPB
#define EPSV 1e-5f

// ws layout (floats): [0, 50331648) total_gated ; [50331648, 52428800) nbr_summed ;
// [52428800, +8) float stats ; then 8 uint counters.
#define NS_OFF 50331648ull
#define ST_OFF 52428800ull

__device__ __forceinline__ uint16_t f2bf(float f) {
  union { float f; uint32_t u; } c; c.f = f;
  uint32_t u = c.u + 0x7FFFu + ((c.u >> 16) & 1u);   // round-to-nearest-even
  return (uint16_t)(u >> 16);
}

union AFrag { uint32_t u[8]; v16bf v; };
union CFrag { float    f[8]; v8f   v; };

__global__ __launch_bounds__(256) void k0_zero(float* ws) {
  float* st = ws + ST_OFF;
  unsigned* cn = (unsigned*)(st + 8);
  if (threadIdx.x < 8) { st[threadIdx.x] = 0.f; cn[threadIdx.x] = 0u; }
}

// GEMM: total_gated[r, c] = (concat_row(r) . W[:, c] + bias[c]) * mask[r, c]
__global__ __launch_bounds__(256) void k1_gemm(
    const float* __restrict__ atom, const float* __restrict__ nbr,
    const float* __restrict__ W,    const float* __restrict__ bias,
    const int*   __restrict__ idx,  const int*   __restrict__ mask,
    float* __restrict__ ws) {
  __shared__ uint16_t sA[RPB * KP];    // 128 rows x (padded K) bf16
  __shared__ uint16_t sW[COLS * KP];   // W transposed: [n][k] bf16
  __shared__ float bsum, bsq;
  __shared__ unsigned bcnt, bcm;

  const int tid = threadIdx.x;
  const int rowBase = blockIdx.x * RPB;

  if (tid == 0) { bsum = 0.f; bsq = 0.f; bcnt = 0u; bcm = 0u; }

  // Stage W^T as bf16 (zero-padded K)
  for (int e = tid; e < COLS * KPAD; e += 256) {
    int n = e / KPAD, k = e - n * KPAD;
    float v = (k < KDIM) ? W[k * COLS + n] : 0.f;
    sW[n * KP + k] = f2bf(v);
  }
  // Stage A rows: [atom_fea(b,n) | atom_fea(b,idx) | nbr_fea(b,n,m) | 0-pad]
  for (int e = tid; e < RPB * KPAD; e += 256) {
    int r = e / KPAD, k = e - r * KPAD;
    int gr = rowBase + r;
    int bn = gr / MM;
    int n  = bn % NN, b = bn / NN;
    float v;
    if (k < FF)               v = atom[(b * NN + n) * FF + k];
    else if (k < 2 * FF)      { int j = idx[gr]; v = atom[(b * NN + j) * FF + (k - FF)]; }
    else if (k < KDIM)        v = nbr[(size_t)gr * NBRF + (k - 2 * FF)];
    else                      v = 0.f;
    sA[r * KP + k] = f2bf(v);
  }
  __syncthreads();

  const int lane = tid & 31;
  const int l15  = lane & 15;
  const int kh   = lane >> 4;
  const int wv   = tid >> 5;     // wave id 0..7
  const int rloc = wv * 16;

  CFrag acc[8];
  #pragma unroll
  for (int t = 0; t < 8; ++t)
    #pragma unroll
    for (int j = 0; j < 8; ++j) acc[t].f[j] = 0.f;

  #pragma unroll
  for (int kk = 0; kk < 6; ++kk) {
    // A fragment: 16x32 bf16 (ISA 16-bit A layout)
    AFrag a;
    #pragma unroll
    for (int v = 0; v < 8; ++v) {
      int k = kk * 32 + (v >> 2) * 16 + kh * 8 + (v & 3) * 2;
      a.u[v] = *(const uint32_t*)&sA[(rloc + l15) * KP + k];
    }
    #pragma unroll
    for (int t = 0; t < 8; ++t) {
      // B fragment: 32x16 bf16 (ISA 16-bit B layout), col = lane&15
      AFrag bf;
      #pragma unroll
      for (int v = 0; v < 8; ++v) {
        int k = kk * 32 + (v >> 2) * 16 + kh * 8 + (v & 3) * 2;
        bf.u[v] = *(const uint32_t*)&sW[(t * 16 + l15) * KP + k];
      }
      acc[t].v = __builtin_amdgcn_wmma_f32_16x16x32_bf16(
          false, a.v, false, bf.v, (short)0, acc[t].v, false, false);
    }
  }

  // Epilogue: bias, mask, store, BN1 stats
  float* gated = ws;
  float lsum = 0.f, lsq = 0.f;
  unsigned lcnt = 0u, lcm = 0u;
  #pragma unroll
  for (int t = 0; t < 8; ++t) {
    int c = t * 16 + l15;
    float bia = bias[c];
    #pragma unroll
    for (int j = 0; j < 8; ++j) {
      int r = rowBase + rloc + j + 8 * kh;     // C/D layout: row = j + 8*khalf
      size_t o = (size_t)r * COLS + c;
      float mv  = (float)mask[o];
      float out = (acc[t].f[j] + bia) * mv;
      gated[o] = out;
      lsum += out; lsq += out * out;
      lcnt += (out != 0.f); lcm += (mv != 0.f);
    }
  }
  atomicAdd(&bsum, lsum); atomicAdd(&bsq, lsq);
  atomicAdd(&bcnt, lcnt); atomicAdd(&bcm, lcm);
  __syncthreads();
  if (tid == 0) {
    float* st = ws + ST_OFF;
    unsigned* cn = (unsigned*)(st + 8);
    atomicAdd(&st[0], bsum); atomicAdd(&st[1], bsq);
    atomicAdd(&cn[0], bcnt); atomicAdd(&cn[1], bcm);
  }
}

// BN1 + sigmoid*softplus gate + sum over M + BN2 stats pass 1
__global__ __launch_bounds__(256) void k2_gate(
    const int* __restrict__ mask, const float* __restrict__ g1,
    const float* __restrict__ b1, float* __restrict__ ws) {
  __shared__ float bsum; __shared__ unsigned bcnt;
  if (threadIdx.x == 0) { bsum = 0.f; bcnt = 0u; }
  __syncthreads();
  const float* gated = ws;
  float* ns = ws + NS_OFF;
  float* st = ws + ST_OFF;
  unsigned* cn = (unsigned*)(st + 8);

  int t  = blockIdx.x * 256 + threadIdx.x;   // 0 .. B*N*F-1
  int f  = t & (FF - 1);
  int bn = t >> 6;

  float c1 = (float)cn[0], cm = (float)cn[1];
  float mu = st[0] / c1;
  float var = (st[1] - 2.f * mu * st[0] + mu * mu * cm) / c1;  // ref var_1
  float inv = rsqrtf(var + EPSV);
  float gA = g1[f], bA = b1[f], gB = g1[FF + f], bB = b1[FF + f];

  size_t base = (size_t)bn * MM * COLS;
  float s = 0.f;
  for (int m = 0; m < MM; ++m) {
    size_t o = base + (size_t)m * COLS;
    float xf = gated[o + f], xc = gated[o + FF + f];
    float hf = ((xf - mu) * inv * gA + bA) * (float)mask[o + f];
    float hc = ((xc - mu) * inv * gB + bB) * (float)mask[o + FF + f];
    float sg = 1.f / (1.f + __expf(-hf));
    float sp = fmaxf(hc, 0.f) + log1pf(__expf(-fabsf(hc)));
    s += sg * sp;
  }
  float mn = (float)mask[base + f];           // m_node = mask[b,n,0,f], f < F
  float v = s * mn;
  ns[t] = v;
  atomicAdd(&bsum, v);
  atomicAdd(&bcnt, (unsigned)(v != 0.f));
  __syncthreads();
  if (threadIdx.x == 0) { atomicAdd(&st[2], bsum); atomicAdd(&cn[2], bcnt); }
}

// BN2 stats pass 2: sum/count of (x - mu2)^2 * m_node
__global__ __launch_bounds__(256) void k3_var2(
    const int* __restrict__ mask, float* __restrict__ ws) {
  __shared__ float bsum; __shared__ unsigned bcnt;
  if (threadIdx.x == 0) { bsum = 0.f; bcnt = 0u; }
  __syncthreads();
  const float* ns = ws + NS_OFF;
  float* st = ws + ST_OFF;
  unsigned* cn = (unsigned*)(st + 8);
  int t  = blockIdx.x * 256 + threadIdx.x;
  int f  = t & (FF - 1);
  int bn = t >> 6;
  float mu2 = st[2] / (float)cn[2];
  float mn = (float)mask[(size_t)bn * MM * COLS + f];
  float d  = ns[t] - mu2;
  float d2 = d * d * mn;
  atomicAdd(&bsum, d2);
  atomicAdd(&bcnt, (unsigned)(d2 != 0.f));
  __syncthreads();
  if (threadIdx.x == 0) { atomicAdd(&st[3], bsum); atomicAdd(&cn[3], bcnt); }
}

// BN2 + residual softplus output
__global__ __launch_bounds__(256) void k4_out(
    const float* __restrict__ atom, const int* __restrict__ mask,
    const float* __restrict__ g2,   const float* __restrict__ b2,
    const float* __restrict__ ws,   float* __restrict__ out) {
  const float* ns = ws + NS_OFF;
  const float* st = ws + ST_OFF;
  const unsigned* cn = (const unsigned*)(st + 8);
  int t  = blockIdx.x * 256 + threadIdx.x;
  int f  = t & (FF - 1);
  int bn = t >> 6;
  float mu2  = st[2] / (float)cn[2];
  float var2 = st[3] / (float)cn[3];
  float inv  = rsqrtf(var2 + EPSV);
  float mn = (float)mask[(size_t)bn * MM * COLS + f];
  float h  = ((ns[t] - mu2) * inv * g2[f] + b2[f]) * mn;
  float x  = atom[t] + h;
  float sp = fmaxf(x, 0.f) + log1pf(__expf(-fabsf(x)));
  out[t] = sp * mn;
}

extern "C" void kernel_launch(void* const* d_in, const int* in_sizes, int n_in,
                              void* d_out, int out_size, void* d_ws, size_t ws_size,
                              hipStream_t stream) {
  (void)in_sizes; (void)n_in; (void)out_size; (void)ws_size;
  const float* atom = (const float*)d_in[0];
  const float* nbr  = (const float*)d_in[1];
  const float* W    = (const float*)d_in[2];
  const float* bias = (const float*)d_in[3];
  const float* g1   = (const float*)d_in[4];
  const float* b1   = (const float*)d_in[5];
  const float* g2   = (const float*)d_in[6];
  const float* b2   = (const float*)d_in[7];
  const int*   idx  = (const int*)d_in[8];
  const int*   mask = (const int*)d_in[9];
  float* out = (float*)d_out;
  float* ws  = (float*)d_ws;

  k0_zero<<<1, 32, 0, stream>>>(ws);
  k1_gemm<<<RTOT / RPB, 256, 0, stream>>>(atom, nbr, W, bias, idx, mask, ws);
  const int nt = BB * NN * FF;
  k2_gate<<<nt / 256, 256, 0, stream>>>(mask, g1, b1, ws);
  k3_var2<<<nt / 256, 256, 0, stream>>>(mask, ws);
  k4_out <<<nt / 256, 256, 0, stream>>>(atom, mask, g2, b2, ws, out);
}
